// MultiHeadAttention_42795054137531
// MI455X (gfx1250) — compile-verified, matrix operand order checked by reading, not run on hardware
//
#include <hip/hip_runtime.h>
#include <hip/hip_bf16.h>

typedef __attribute__((ext_vector_type(16))) _Float16 v16h;
typedef __attribute__((ext_vector_type(8)))  _Float16 v8h;
typedef __attribute__((ext_vector_type(8)))  float    v8f;

#define B_  2
#define S_  2048
#define DM_ 1024
#define H_  16
#define DH_ 64

#if defined(__has_builtin)
#  if __has_builtin(__builtin_amdgcn_tensor_load_to_lds) && \
      __has_builtin(__builtin_amdgcn_s_wait_tensorcnt)
#    define USE_TDM 1
#  endif
#endif
#ifndef USE_TDM
#  define USE_TDM 0
#endif

__device__ __forceinline__ v8f wmma32(v16h a, v16h b, v8f c) {
  return __builtin_amdgcn_wmma_f32_16x16x32_f16(
      /*neg_a=*/false, a, /*neg_b=*/false, b,
      /*c_mod=*/(short)0, c, /*reuse_a=*/false, /*reuse_b=*/false);
}

// ---- fragment loader ------------------------------------------------------
// ISA 16-bit A layout: lane<16 holds K 0..7,16..23 ; lane>=16 holds K 8..15,24..31.
// Per lane: two contiguous 16-byte runs at col offsets {o, o+16}, o=(lane>>4)*8.
__device__ __forceinline__ v16h load_fragA_f16(const _Float16* __restrict__ base,
                                               int ldm, int lane) {
  const _Float16* row = base + (lane & 15) * ldm + ((lane >> 4) << 3);
  v8h lo = *(const v8h*)(row);
  v8h hi = *(const v8h*)(row + 16);
  v16h out;
#pragma unroll
  for (int i = 0; i < 8; ++i) { out[i] = lo[i]; out[8 + i] = hi[i]; }
  return out;
}

// ---- Tensor Data Mover ----------------------------------------------------
#if USE_TDM
typedef __attribute__((ext_vector_type(4))) unsigned int u32x4;
typedef __attribute__((ext_vector_type(8))) int          i32x8;
typedef __attribute__((ext_vector_type(4))) int          i32x4;

__device__ __forceinline__ unsigned lds_off(const void* p) {
  // flat shared address low 32 bits == workgroup-relative LDS byte offset
  return (unsigned)(size_t)p;
}

// 2-D f16 tile load: tile (tile0 x tile1) from global (row stride stride0
// elements) into contiguous LDS. Tensor dims == tile dims (never OOB).
__device__ __forceinline__ void tdm_load_2d_f16(unsigned lds_byte, const void* gptr,
                                                unsigned tile0, unsigned tile1,
                                                unsigned stride0) {
  unsigned long long ga = (unsigned long long)(size_t)gptr;
  u32x4 g0;
  g0[0] = 1u;                                               // count=1, user D#
  g0[1] = lds_byte;                                         // lds_addr
  g0[2] = (unsigned)ga;                                     // global_addr[31:0]
  g0[3] = ((unsigned)(ga >> 32) & 0x1FFFFFFu) | (2u << 30); // addr[56:32] | type=2
  i32x8 g1;
  g1[0] = (int)(1u << 16);                                  // data_size=1 (2 bytes)
  g1[1] = (int)((tile0 & 0xFFFFu) << 16);                   // tensor_dim0 lo16
  g1[2] = (int)((tile1 & 0xFFFFu) << 16);                   // tensor_dim1 lo16
  g1[3] = (int)((tile0 & 0xFFFFu) << 16);                   // tile_dim0
  g1[4] = (int)(tile1 & 0xFFFFu);                           // tile_dim1 | tile_dim2=0
  g1[5] = (int)stride0;                                     // tensor_dim0_stride lo32
  g1[6] = 0;
  g1[7] = 0;
  i32x4 z4 = {0, 0, 0, 0};
#if __clang_major__ >= 23
  i32x8 z8 = {0, 0, 0, 0, 0, 0, 0, 0};
  __builtin_amdgcn_tensor_load_to_lds(g0, g1, z4, z4, z8, 0);
#else
  __builtin_amdgcn_tensor_load_to_lds(g0, g1, z4, z4, 0);
#endif
}
#endif  // USE_TDM

// ---------------------------------------------------------------------------
// Kernel 0a: weight prep.  wt[z][n][k] = (f16) W_z[k][n]  (B^T layout, f16).
// ---------------------------------------------------------------------------
__global__ __launch_bounds__(256) void k_prep_w(
    const float* __restrict__ Wq, const float* __restrict__ Wk,
    const float* __restrict__ Wv, const float* __restrict__ Wo,
    _Float16* __restrict__ wt) {
  const float* src = (blockIdx.y == 0) ? Wq
                   : (blockIdx.y == 1) ? Wk
                   : (blockIdx.y == 2) ? Wv : Wo;
  _Float16* dst = wt + (size_t)blockIdx.y * DM_ * DM_;
  const int idx = blockIdx.x * 256 + threadIdx.x;
  const int n = idx >> 10, k = idx & (DM_ - 1);
  dst[(size_t)n * DM_ + k] = (_Float16)src[(size_t)k * DM_ + n];
}

// ---------------------------------------------------------------------------
// Kernel 0b: activation prep.  xh[z][t][k] = (f16) X_z[t][k].
// ---------------------------------------------------------------------------
__global__ __launch_bounds__(256) void k_prep_x(
    const float* __restrict__ Qin, const float* __restrict__ Kin,
    const float* __restrict__ Vin, _Float16* __restrict__ xh) {
  const float* src = (blockIdx.y == 0) ? Qin : (blockIdx.y == 1) ? Kin : Vin;
  _Float16* dst = xh + (size_t)blockIdx.y * B_ * S_ * DM_;
  const size_t idx = (size_t)blockIdx.x * 256 + threadIdx.x;
  dst[idx] = (_Float16)src[idx];
}

// ---------------------------------------------------------------------------
// Kernel 1: QKV projection as TDM-fed LDS GEMM.
// Block = 256 threads (8 waves) computes a 64x128 tile of X @ W.
// Per k-step: TDM stages X panel (64x32) + W^T panel (128x32) into LDS,
// double-buffered; each wave does 4 WMMAs on shared operands.
// q,k -> [B,H,S,DH] f16 ; v -> [B,H,DH,S] f16.
// ---------------------------------------------------------------------------
__global__ __launch_bounds__(256) void k_qkv(
    const _Float16* __restrict__ xh, const _Float16* __restrict__ wt,
    const float* __restrict__ bq, const float* __restrict__ bk,
    const float* __restrict__ bv,
    _Float16* __restrict__ qh, _Float16* __restrict__ kh,
    _Float16* __restrict__ vt) {
  __shared__ __align__(16) _Float16 xbuf[2][64][32];    // X panel  (m, k)
  __shared__ __align__(16) _Float16 wbuf[2][128][32];   // W^T panel (n, k)

  const int lane = threadIdx.x & 31;
  const int wave = threadIdx.x >> 5;
  const int z = blockIdx.z;

  const _Float16* X    = xh + (size_t)z * B_ * S_ * DM_;
  const _Float16* wtz  = wt + (size_t)z * DM_ * DM_;
  const float*    bias = (z == 0) ? bq : ((z == 1) ? bk : bv);

  const int nblk = blockIdx.x * 128;          // 128 output cols per block
  const int mblk = blockIdx.y * 64;           // 64 rows per block
  const int mw   = (wave & 3) * 16;           // wave's m-subtile in panel
  const int nw   = (wave >> 2) * 64;          // wave's n-group in panel

  v8f acc[4];
#pragma unroll
  for (int t = 0; t < 4; ++t) acc[t] = (v8f){};

#if USE_TDM
  if (wave == 0) {
    tdm_load_2d_f16(lds_off(&xbuf[0][0][0]), X + (size_t)mblk * DM_, 32, 64, DM_);
    tdm_load_2d_f16(lds_off(&wbuf[0][0][0]), wtz + (size_t)nblk * DM_, 32, 128, DM_);
  }
#endif

  for (int kb = 0; kb < DM_; kb += 32) {
    const int cur = (kb >> 5) & 1;
#if USE_TDM
    if (wave == 0) {
      if (kb + 32 < DM_) {
        tdm_load_2d_f16(lds_off(&xbuf[cur ^ 1][0][0]),
                        X + (size_t)mblk * DM_ + kb + 32, 32, 64, DM_);
        tdm_load_2d_f16(lds_off(&wbuf[cur ^ 1][0][0]),
                        wtz + (size_t)nblk * DM_ + kb + 32, 32, 128, DM_);
        __builtin_amdgcn_s_wait_tensorcnt(2);
      } else {
        __builtin_amdgcn_s_wait_tensorcnt(0);
      }
    }
    __syncthreads();
#else
    for (int i = threadIdx.x; i < (64 * 32) / 8; i += 256) {
      const int row = i >> 2, c = (i & 3) * 8;
      *(v8h*)&xbuf[cur][row][c] =
          *(const v8h*)(X + (size_t)(mblk + row) * DM_ + kb + c);
    }
    for (int i = threadIdx.x; i < (128 * 32) / 8; i += 256) {
      const int row = i >> 2, c = (i & 3) * 8;
      *(v8h*)&wbuf[cur][row][c] =
          *(const v8h*)(wtz + (size_t)(nblk + row) * DM_ + kb + c);
    }
    __syncthreads();
#endif

    v16h a = load_fragA_f16(&xbuf[cur][mw][0], 32, lane);
#pragma unroll
    for (int t = 0; t < 4; ++t) {
      v16h b = load_fragA_f16(&wbuf[cur][nw + t * 16][0], 32, lane);
      acc[t] = wmma32(a, b, acc[t]);
    }
    __syncthreads();
  }

  const int nl = lane & 15;
  const int mb = mblk + mw + ((lane >> 4) << 3);
#pragma unroll
  for (int t = 0; t < 4; ++t) {
    const int nc = nblk + nw + t * 16 + nl;
    const float bv_ = bias[nc];
    const int h = nc >> 6;
    const int d = nc & (DH_ - 1);
#pragma unroll
    for (int r = 0; r < 8; ++r) {
      const int tok = mb + r;
      const int bb = tok >> 11;
      const int s  = tok & (S_ - 1);
      const _Float16 val = (_Float16)(acc[t][r] + bv_);
      if (z == 2)
        vt[(((size_t)bb * H_ + h) * DH_ + d) * S_ + s] = val;   // [B,H,DH,S]
      else if (z == 1)
        kh[(((size_t)bb * H_ + h) * S_ + s) * DH_ + d] = val;   // [B,H,S,DH]
      else
        qh[(((size_t)bb * H_ + h) * S_ + s) * DH_ + d] = val;
    }
  }
}

// ---------------------------------------------------------------------------
// Kernel 2: flash attention, TDM-staged K/V strips, double-buffered in LDS.
// grid (S/64, B*H), block 128 (4 waves; one 16-row q-tile per wave).
// ---------------------------------------------------------------------------
__global__ __launch_bounds__(128) void k_attn(
    const _Float16* __restrict__ qh, const _Float16* __restrict__ kh,
    const _Float16* __restrict__ vt, const unsigned char* __restrict__ mask,
    _Float16* __restrict__ ctxh) {
  __shared__ __align__(16) _Float16 kbuf[2][32][DH_];   // key strip  (kpos, d)
  __shared__ __align__(16) _Float16 vbuf[2][DH_][32];   // value strip (d, kpos)
  __shared__ __align__(16) _Float16 pstage[4][16][40];  // per-wave P tile

  const int lane = threadIdx.x & 31;
  const int wave = threadIdx.x >> 5;
  const int bh = blockIdx.y;
  const int b = bh >> 4, h = bh & 15;
  const int q0 = (blockIdx.x * 4 + wave) * 16;

  const _Float16* qbase = qh + (size_t)bh * S_ * DH_;
  const _Float16* kbase = kh + (size_t)bh * S_ * DH_;
  const _Float16* vbase = vt + (size_t)bh * DH_ * S_;
  const unsigned char* mbase = mask + (size_t)b * S_ * S_;

  const v16h qa0 = load_fragA_f16(qbase + (size_t)q0 * DH_, DH_, lane);
  const v16h qa1 = load_fragA_f16(qbase + (size_t)q0 * DH_ + 32, DH_, lane);

  float mrun[8], lrun[8];
  v8f acc[4];
#pragma unroll
  for (int r = 0; r < 8; ++r) { mrun[r] = -1e30f; lrun[r] = 0.0f; }
#pragma unroll
  for (int t = 0; t < 4; ++t) acc[t] = (v8f){};

  const int mb = (lane >> 4) << 3;
  const int nl = lane & 15;
  const float scale = 0.125f;   // 1/sqrt(DH)

#if USE_TDM
  if (wave == 0) {
    tdm_load_2d_f16(lds_off(&kbuf[0][0][0]), kbase, DH_, 32, DH_);
    tdm_load_2d_f16(lds_off(&vbuf[0][0][0]), vbase, 32, DH_, S_);
  }
#endif

  for (int kt = 0; kt < S_; kt += 32) {
    const int cur = (kt >> 5) & 1;
#if USE_TDM
    if (wave == 0) {
      if (kt + 32 < S_) {
        tdm_load_2d_f16(lds_off(&kbuf[cur ^ 1][0][0]),
                        kbase + (size_t)(kt + 32) * DH_, DH_, 32, DH_);
        tdm_load_2d_f16(lds_off(&vbuf[cur ^ 1][0][0]),
                        vbase + (kt + 32), 32, DH_, S_);
        __builtin_amdgcn_s_wait_tensorcnt(2);   // current strip complete
      } else {
        __builtin_amdgcn_s_wait_tensorcnt(0);
      }
    }
    __syncthreads();
#else
    {
      const _Float16* gk = kbase + (size_t)kt * DH_;
      for (int i = threadIdx.x; i < (32 * DH_) / 8; i += 128)
        *((v8h*)&kbuf[cur][0][0] + i) = *((const v8h*)gk + i);
      for (int i = threadIdx.x; i < (DH_ * 32) / 8; i += 128) {
        const int row = i >> 2, c = (i & 3) * 8;
        *(v8h*)&vbuf[cur][row][c] = *(const v8h*)(vbase + (size_t)row * S_ + kt + c);
      }
    }
    __syncthreads();
#endif

    // ---- scores: two 16x16 tiles over key strip [kt, kt+32) -------------
    v16h kb0a = load_fragA_f16(&kbuf[cur][0][0], DH_, lane);
    v16h kb0b = load_fragA_f16(&kbuf[cur][0][0] + 32, DH_, lane);
    v16h kb1a = load_fragA_f16(&kbuf[cur][16][0], DH_, lane);
    v16h kb1b = load_fragA_f16(&kbuf[cur][16][0] + 32, DH_, lane);
    v8f s0 = {}, s1 = {};
    s0 = wmma32(qa0, kb0a, s0);
    s0 = wmma32(qa1, kb0b, s0);
    s1 = wmma32(qa0, kb1a, s1);
    s1 = wmma32(qa1, kb1b, s1);

    // ---- online softmax over the 32 columns (half-wave row reductions) --
#pragma unroll
    for (int r = 0; r < 8; ++r) {
      const int row = q0 + mb + r;
      const unsigned char* mm = mbase + (size_t)row * S_ + kt + nl;
      float v0 = s0[r] * scale; if (mm[0])  v0 = -1e9f;
      float v1 = s1[r] * scale; if (mm[16]) v1 = -1e9f;

      float mx = fmaxf(v0, v1);
      mx = fmaxf(mx, __shfl_xor(mx, 1, 16));
      mx = fmaxf(mx, __shfl_xor(mx, 2, 16));
      mx = fmaxf(mx, __shfl_xor(mx, 4, 16));
      mx = fmaxf(mx, __shfl_xor(mx, 8, 16));

      const float mnew  = fmaxf(mrun[r], mx);
      const float alpha = __expf(mrun[r] - mnew);
      const float e0 = __expf(v0 - mnew);
      const float e1 = __expf(v1 - mnew);
      float rs = e0 + e1;
      rs += __shfl_xor(rs, 1, 16);
      rs += __shfl_xor(rs, 2, 16);
      rs += __shfl_xor(rs, 4, 16);
      rs += __shfl_xor(rs, 8, 16);
      lrun[r] = lrun[r] * alpha + rs;
      mrun[r] = mnew;
#pragma unroll
      for (int t = 0; t < 4; ++t) acc[t][r] *= alpha;

      pstage[wave][mb + r][nl]      = (_Float16)e0;
      pstage[wave][mb + r][nl + 16] = (_Float16)e1;
    }

    // ---- P(16x32) @ V(32x64); V^T rows are contiguous in vbuf -----------
    v16h pa = load_fragA_f16(&pstage[wave][0][0], 40, lane);
#pragma unroll
    for (int t = 0; t < 4; ++t) {
      v16h vb = load_fragA_f16(&vbuf[cur][t * 16][0], 32, lane);
      acc[t] = wmma32(pa, vb, acc[t]);
    }
    __syncthreads();   // all reads of buf[cur] done before it is refilled
  }

  // ---- normalize + store ctx f16 in [B, S, H*DH] -------------------------
#pragma unroll
  for (int t = 0; t < 4; ++t) {
#pragma unroll
    for (int r = 0; r < 8; ++r) {
      const int srow = q0 + mb + r;
      const float val = acc[t][r] / lrun[r];
      ctxh[((size_t)b * S_ + srow) * DM_ + h * DH_ + t * 16 + nl] = (_Float16)val;
    }
  }
}

// ---------------------------------------------------------------------------
// Kernel 3: out-projection + bias + residual + LayerNorm.
// grid (B*S/16), block 256 (8 waves x 8 column tiles = 1024 cols).
// ---------------------------------------------------------------------------
__global__ __launch_bounds__(256) void k_outln(
    const _Float16* __restrict__ ctxh, const _Float16* __restrict__ wto,
    const float* __restrict__ bo, const float* __restrict__ Qin,
    const float* __restrict__ gamma, const float* __restrict__ beta,
    float* __restrict__ out) {
  __shared__ float rows[16][DM_];   // 64 KB: 16 full output rows

  const int lane = threadIdx.x & 31;
  const int wave = threadIdx.x >> 5;
  const int m0 = blockIdx.x * 16;

  v8f acc[8];
#pragma unroll
  for (int t = 0; t < 8; ++t) acc[t] = (v8f){};

  for (int kb = 0; kb < DM_; kb += 32) {
    v16h a = load_fragA_f16(ctxh + (size_t)m0 * DM_ + kb, DM_, lane);
#pragma unroll
    for (int t = 0; t < 8; ++t) {
      const int n0 = (wave * 8 + t) * 16;
      __builtin_prefetch(wto + (size_t)n0 * DM_ + kb + 32, 0, 0);
      v16h bfr = load_fragA_f16(wto + (size_t)n0 * DM_ + kb, DM_, lane);
      acc[t] = wmma32(a, bfr, acc[t]);
    }
  }

  const int mb = (lane >> 4) << 3;
  const int nl = lane & 15;
#pragma unroll
  for (int t = 0; t < 8; ++t) {
    const int n0 = (wave * 8 + t) * 16;
#pragma unroll
    for (int r = 0; r < 8; ++r) rows[mb + r][n0 + nl] = acc[t][r];
  }
  __syncthreads();

  const int r  = threadIdx.x >> 4;
  const int c0 = threadIdx.x & 15;
  const size_t grow = (size_t)(m0 + r) * DM_;

  float sum = 0.0f, sq = 0.0f;
#pragma unroll 4
  for (int i = 0; i < 64; ++i) {
    const int c = c0 + i * 16;
    const float x = rows[r][c] + bo[c] + Qin[grow + c];
    rows[r][c] = x;
    sum += x;
    sq  += x * x;
  }
  sum += __shfl_xor(sum, 1, 16); sq += __shfl_xor(sq, 1, 16);
  sum += __shfl_xor(sum, 2, 16); sq += __shfl_xor(sq, 2, 16);
  sum += __shfl_xor(sum, 4, 16); sq += __shfl_xor(sq, 4, 16);
  sum += __shfl_xor(sum, 8, 16); sq += __shfl_xor(sq, 8, 16);

  const float mu  = sum * (1.0f / DM_);
  const float var = sq * (1.0f / DM_) - mu * mu;
  const float inv = rsqrtf(var + 1e-5f);

#pragma unroll 4
  for (int i = 0; i < 64; ++i) {
    const int c = c0 + i * 16;
    out[grow + c] = (rows[r][c] - mu) * inv * gamma[c] + beta[c];
  }
}

// ---------------------------------------------------------------------------
extern "C" void kernel_launch(void* const* d_in, const int* in_sizes, int n_in,
                              void* d_out, int out_size, void* d_ws, size_t ws_size,
                              hipStream_t stream) {
  const float* Q = (const float*)d_in[0];
  const float* K = (const float*)d_in[1];
  const float* V = (const float*)d_in[2];
  const unsigned char* mask = (const unsigned char*)d_in[3];
  const float* Wq = (const float*)d_in[4];
  const float* bq = (const float*)d_in[5];
  const float* Wk = (const float*)d_in[6];
  const float* bk = (const float*)d_in[7];
  const float* Wv = (const float*)d_in[8];
  const float* bv = (const float*)d_in[9];
  const float* Wo = (const float*)d_in[10];
  const float* bo = (const float*)d_in[11];
  const float* gamma = (const float*)d_in[12];
  const float* beta  = (const float*)d_in[13];
  float* out = (float*)d_out;

  const size_t tok = (size_t)B_ * S_ * DM_;   // 4M f16 elems per tensor
  _Float16* qh   = (_Float16*)d_ws;
  _Float16* kh   = qh + tok;
  _Float16* vt   = kh + tok;                  // value stored [B,H,DH,S]
  _Float16* ctxh = vt + tok;
  _Float16* wt   = ctxh + tok;                // 4 x [DM,DM] f16 transposed weights
  _Float16* xh   = wt + 4 * (size_t)DM_ * DM_; // 3 x [B*S,DM] f16 activations

  k_prep_w<<<dim3((DM_ * DM_) / 256, 4), 256, 0, stream>>>(Wq, Wk, Wv, Wo, wt);
  k_prep_x<<<dim3((unsigned)(tok / 256), 3), 256, 0, stream>>>(Q, K, V, xh);

  dim3 g1(DM_ / 128, (B_ * S_) / 64, 3);
  k_qkv<<<g1, 256, 0, stream>>>(xh, wt, bq, bk, bv, qh, kh, vt);

  dim3 g2(S_ / 64, B_ * H_);
  k_attn<<<g2, 128, 0, stream>>>(qh, kh, vt, mask, ctxh);

  k_outln<<<(B_ * S_) / 16, 256, 0, stream>>>(ctxh, wt + 3 * (size_t)DM_ * DM_,
                                              bo, Q, gamma, beta, out);
}
